// SSM_7232724927260
// MI455X (gfx1250) — matrix operand order repaired and checked by reading, not compiled, not run on hardware
//
#include <hip/hip_runtime.h>

#define CPD       32
#define DIM       1024
#define BATCH     16
#define FRAMES    2048
#define HOP       512
#define KBLOCKS   33            // 32 state K-blocks + 1 control/PB block
#define LROW      1056          // 1024 state cols + 32 ctrl cols per LDS row
#define NWG       8             // workgroups in persistent scan kernel
#define SCAN_THREADS 128        // 4 waves; 2 tiles/wave -> 64 tiles = N/16

// ---- CDNA5 async global->LDS path (probe-guarded) ----
#if defined(__has_builtin)
#if __has_builtin(__builtin_amdgcn_global_load_async_to_lds_b128)
#define HAVE_ASYNC_LDS 1
#endif
#if __has_builtin(__builtin_amdgcn_s_wait_asynccnt)
#define HAVE_WAIT_ASYNC 1
#endif
#endif
#ifndef HAVE_ASYNC_LDS
#define HAVE_ASYNC_LDS 0
#endif
#ifndef HAVE_WAIT_ASYNC
#define HAVE_WAIT_ASYNC 0
#endif

typedef __attribute__((ext_vector_type(16))) __bf16 v16bf;
typedef __attribute__((ext_vector_type(8)))  float  v8f;
typedef int v4i_t __attribute__((ext_vector_type(4)));
typedef __attribute__((address_space(1))) v4i_t* gv4_p;   // global v4i*
typedef __attribute__((address_space(3))) v4i_t* lv4_p;   // LDS v4i*

union Frag16 { v16bf v; uint4 q[2]; unsigned short u[16]; };
union Frag8f { v8f v; float f[8]; };

__device__ __forceinline__ unsigned short f2bf(float x) {
  unsigned int u = __float_as_uint(x);
  unsigned int r = 0x7FFFu + ((u >> 16) & 1u);
  return (unsigned short)((u + r) >> 16);
}

__device__ __forceinline__ v8f wmma_bf16(const Frag16& a, const Frag16& b, v8f c) {
  return __builtin_amdgcn_wmma_f32_16x16x32_bf16(false, a.v, false, b.v,
                                                 (short)0, c, false, false);
}

// B operand: packed so each lane reads 16 contiguous bf16 (32B)
__device__ __forceinline__ void load_b(Frag16& b, const unsigned short* Bpk,
                                       int kb, int ncol, int kbB) {
  const unsigned short* p = Bpk + ((unsigned)(kb * DIM + ncol) << 5) + kbB;
  b.q[0] = *(const uint4*)p;
  b.q[1] = *(const uint4*)(p + 8);
}
// A operand from LDS: two 16B runs per ISA 7.12.2 (K{0-7,16-23}/{8-15,24-31})
__device__ __forceinline__ void load_a_lds(Frag16& a, const unsigned short* lds,
                                           int m, int kb, int kbA) {
  const unsigned short* p = lds + m * LROW + kb * 32 + kbA;
  a.q[0] = *(const uint4*)p;
  a.q[1] = *(const uint4*)(p + 16);
}
// A operand from a global bf16 row (row stride 1024)
__device__ __forceinline__ void load_a_glb(Frag16& a, const unsigned short* row,
                                           int kb, int kbA) {
  const unsigned short* p = row + kb * 32 + kbA;
  a.q[0] = *(const uint4*)p;
  a.q[1] = *(const uint4*)(p + 16);
}

// ---------------------------------------------------------------- init ----
__global__ void k_init_zero(float* __restrict__ wsHead, unsigned nWs) {
  unsigned i = blockIdx.x * blockDim.x + threadIdx.x;
  unsigned stride = gridDim.x * blockDim.x;
  for (unsigned j = i; j < nWs; j += stride) wsHead[j] = 0.f;
}

// ------------------------------------------- PB = proj@IM, PD = proj@DM ---
__global__ void k_pbpd(const float* __restrict__ proj,
                       const float* __restrict__ im,
                       const float* __restrict__ dm,
                       float* __restrict__ PB, float* __restrict__ PD) {
  int idx = blockIdx.x * blockDim.x + threadIdx.x;   // 0..65535
  int which = idx >> 15;
  int r = idx & 32767;
  int c = r >> 10, s = r & 1023;
  const float* M = which ? dm : im;
  float acc = 0.f;
  for (int i = 0; i < DIM; ++i) acc += proj[c * DIM + i] * M[i * DIM + s];
  (which ? PD : PB)[c * DIM + s] = acc;
}

// ---- pack [[A],[PBsrc]] (K=1056 x N=1024) into bf16 B-fragment layout ----
__global__ void k_pack(const float* __restrict__ A,
                       const float* __restrict__ PBsrc,
                       unsigned short* __restrict__ dst) {
  int idx = blockIdx.x * blockDim.x + threadIdx.x;   // 33*1024*32 threads
  int t  = idx & 31;
  int n  = (idx >> 5) & 1023;
  int kb = idx >> 15;
  int kk = (t & 7) | ((t & 8) << 1) | ((t & 16) >> 1);
  float v = (kb < 32) ? A[(kb * 32 + kk) * DIM + n] : PBsrc[kk * DIM + n];
  dst[idx] = f2bf(v);
}

// ------------------------------------------------ sequential scan kernel --
// S_all[f] = [S_all[f-1] | ctrl_f](16x1056) @ BpkA ; persistent, 1 barrier/frame
__global__ void k_scan(const float* __restrict__ control,
                       const unsigned short* __restrict__ BpkA,
                       const unsigned short* __restrict__ Szero,
                       unsigned short* __restrict__ S_all,
                       unsigned* __restrict__ sync) {
  __shared__ unsigned short s_lds[BATCH * LROW];   // 33 KB
  const int tid  = threadIdx.x;
  const int lane = tid & 31;
  const int wid  = tid >> 5;
  const int m    = lane & 15;
  const int kbA  = (lane & 16) ? 8 : 0;
  const int kbB  = (lane & 16) ? 16 : 0;
  const int base = (blockIdx.x * 4 + wid) * 32;    // 2 tiles per wave
  const int nc0  = base + m;
  const int nc1  = base + 16 + m;
  unsigned* cnt = sync;
  unsigned* gen = sync + 1;

#pragma unroll 1
  for (int f = 0; f < FRAMES; ++f) {
    const unsigned short* Sprev =
        f ? (S_all + (unsigned)(f - 1) * (BATCH * DIM)) : Szero;

    // stage previous state (bf16, rows 2048B) into LDS rows of LROW
    for (int ch = tid; ch < 2048; ch += SCAN_THREADS) {
      int row = ch >> 7, c16 = ch & 127;
      const unsigned short* g = Sprev + (row << 10) + (c16 << 3);
      unsigned short* l = s_lds + row * LROW + (c16 << 3);
#if HAVE_ASYNC_LDS
      __builtin_amdgcn_global_load_async_to_lds_b128((gv4_p)g, (lv4_p)l, 0, 0);
#else
      *(uint4*)l = *(const uint4*)g;
#endif
    }
    // control column (f32 -> bf16) into trailing 32 cols of each LDS row
    for (int i = tid; i < BATCH * CPD; i += SCAN_THREADS) {
      int b = i >> 5, c = i & 31;
      s_lds[b * LROW + DIM + c] = f2bf(control[(b * CPD + c) * FRAMES + f]);
    }
#if HAVE_ASYNC_LDS
#if HAVE_WAIT_ASYNC
    __builtin_amdgcn_s_wait_asynccnt(0);
#else
    asm volatile("s_wait_asynccnt 0x0" ::: "memory");
#endif
#endif
    __syncthreads();

    Frag8f acc0, acc1;
#pragma unroll
    for (int r = 0; r < 8; ++r) { acc0.f[r] = 0.f; acc1.f[r] = 0.f; }

    // software-pipelined K loop: 33 blocks, unroll-by-2, two register sets
    Frag16 aA, bA0, bA1, aB, bB0, bB1;
    load_a_lds(aA, s_lds, m, 0, kbA);
    load_b(bA0, BpkA, 0, nc0, kbB);
    load_b(bA1, BpkA, 0, nc1, kbB);
#pragma unroll 1
    for (int kb = 0; kb < 32; kb += 2) {
      load_a_lds(aB, s_lds, m, kb + 1, kbA);
      load_b(bB0, BpkA, kb + 1, nc0, kbB);
      load_b(bB1, BpkA, kb + 1, nc1, kbB);
      acc0.v = wmma_bf16(aA, bA0, acc0.v);
      acc1.v = wmma_bf16(aA, bA1, acc1.v);
      load_a_lds(aA, s_lds, m, kb + 2, kbA);        // kb+2 <= 32
      load_b(bA0, BpkA, kb + 2, nc0, kbB);
      load_b(bA1, BpkA, kb + 2, nc1, kbB);
      acc0.v = wmma_bf16(aB, bB0, acc0.v);
      acc1.v = wmma_bf16(aB, bB1, acc1.v);
    }
    acc0.v = wmma_bf16(aA, bA0, acc0.v);            // block 32 (ctrl/PB)
    acc1.v = wmma_bf16(aA, bA1, acc1.v);

    unsigned short* Sf = S_all + (unsigned)f * (BATCH * DIM);
#pragma unroll
    for (int r = 0; r < 8; ++r) {
      int M = r + ((lane >> 4) << 3);
      Sf[(M << 10) + nc0] = f2bf(acc0.f[r]);
      Sf[(M << 10) + nc1] = f2bf(acc1.f[r]);
    }

    // ---- grid barrier (release all stores, then count + spin) ----
    __threadfence();
    __syncthreads();
    if (tid == 0) {
      unsigned prev = __hip_atomic_fetch_add(cnt, 1u, __ATOMIC_ACQ_REL,
                                             __HIP_MEMORY_SCOPE_AGENT);
      if (prev + 1u == NWG) {
        __hip_atomic_store(cnt, 0u, __ATOMIC_RELAXED, __HIP_MEMORY_SCOPE_AGENT);
        __hip_atomic_fetch_add(gen, 1u, __ATOMIC_ACQ_REL,
                               __HIP_MEMORY_SCOPE_AGENT);
      } else {
        while (__hip_atomic_load(gen, __ATOMIC_ACQUIRE,
                                 __HIP_MEMORY_SCOPE_AGENT) < (unsigned)(f + 1))
          __builtin_amdgcn_s_sleep(1);
      }
    }
    __syncthreads();
  }
}

// ---------------- bulk output GEMM + fused overlap-add (no atomics) -------
// region (f, jt): d_out[b, f*512+jt*16+c] = Tile(f, jt)[b,c] + Tile(f-1, jt+32)[b,c]
__global__ void k_output(const float* __restrict__ control,
                         const unsigned short* __restrict__ S_all,
                         const unsigned short* __restrict__ Szero,
                         const unsigned short* __restrict__ BpkC,
                         float* __restrict__ out) {
  const int lane = threadIdx.x & 31;
  const int wid  = threadIdx.x >> 5;
  const int gw   = blockIdx.x * 8 + wid;   // 65536 waves total
  const int f    = gw >> 5;                // frame
  const int jt   = gw & 31;                // 16-col block within hop
  const int m    = lane & 15;              // A row == batch b
  const int kbA  = (lane & 16) ? 8 : 0;
  const int kbB  = (lane & 16) ? 16 : 0;
  const int nc0  = jt * 16 + m;            // first-term tile column (<512)
  const int nc1  = nc0 + HOP;              // second-term tile column (>=512)

  const unsigned short* Arow0 = S_all + ((unsigned)(f * BATCH + m) << 10);
  const unsigned short* Arow1 =
      (f > 0) ? (S_all + ((unsigned)((f - 1) * BATCH + m) << 10))
              : (Szero + (m << 10));

  Frag8f acc0, acc1;
#pragma unroll
  for (int r = 0; r < 8; ++r) { acc0.f[r] = 0.f; acc1.f[r] = 0.f; }

  Frag16 aA0, aA1, bA0, bA1, aB0, aB1, bB0, bB1;
  load_a_glb(aA0, Arow0, 0, kbA);
  load_a_glb(aA1, Arow1, 0, kbA);
  load_b(bA0, BpkC, 0, nc0, kbB);
  load_b(bA1, BpkC, 0, nc1, kbB);
#pragma unroll 1
  for (int kb = 0; kb < 32; kb += 2) {
    load_a_glb(aB0, Arow0, kb + 1, kbA);
    load_a_glb(aB1, Arow1, kb + 1, kbA);
    load_b(bB0, BpkC, kb + 1, nc0, kbB);
    load_b(bB1, BpkC, kb + 1, nc1, kbB);
    acc0.v = wmma_bf16(aA0, bA0, acc0.v);
    acc1.v = wmma_bf16(aA1, bA1, acc1.v);
    if (kb + 2 < 32) {
      load_a_glb(aA0, Arow0, kb + 2, kbA);
      load_a_glb(aA1, Arow1, kb + 2, kbA);
      load_b(bA0, BpkC, kb + 2, nc0, kbB);
      load_b(bA1, BpkC, kb + 2, nc1, kbB);
    }
    acc0.v = wmma_bf16(aB0, bB0, acc0.v);
    acc1.v = wmma_bf16(aB1, bB1, acc1.v);
  }
  // block 32: control channels (K index == channel c)
  Frag16 ac0, ac1, bc0, bc1;
#pragma unroll
  for (int j = 0; j < 16; ++j) {
    int c = (j < 8) ? (kbA + j) : (kbA + 8 + j);
    ac0.u[j] = f2bf(control[(m * CPD + c) * FRAMES + f]);
    ac1.u[j] = (f > 0) ? f2bf(control[(m * CPD + c) * FRAMES + (f - 1)])
                       : (unsigned short)0;
  }
  load_b(bc0, BpkC, 32, nc0, kbB);
  load_b(bc1, BpkC, 32, nc1, kbB);
  acc0.v = wmma_bf16(ac0, bc0, acc0.v);
  acc1.v = wmma_bf16(ac1, bc1, acc1.v);

  // each d_out element written exactly once; coalesced 64B per half-wave
#pragma unroll
  for (int r = 0; r < 8; ++r) {
    int b = r + ((lane >> 4) << 3);
    out[(unsigned)b * (FRAMES * HOP) + (unsigned)f * HOP + nc0] =
        acc0.f[r] + acc1.f[r];
  }
}

// ------------------------------------------------------------------ host --
extern "C" void kernel_launch(void* const* d_in, const int* in_sizes, int n_in,
                              void* d_out, int out_size, void* d_ws, size_t ws_size,
                              hipStream_t stream) {
  const float* control = (const float*)d_in[0];
  const float* proj    = (const float*)d_in[1];
  const float* smat    = (const float*)d_in[2];
  const float* imat    = (const float*)d_in[3];
  const float* omat    = (const float*)d_in[4];
  const float* dmat    = (const float*)d_in[5];
  float* out = (float*)d_out;

  char* ws = (char*)d_ws;
  unsigned*       sync  = (unsigned*)      (ws + 0);        // 2 uints
  unsigned short* Szero = (unsigned short*)(ws + 256);      // 16*1024 bf16 (zeros)
  float*          PB    = (float*)         (ws + 33024);    // 32*1024 f32
  float*          PD    = (float*)         (ws + 164096);   // 32*1024 f32
  unsigned short* BpkA  = (unsigned short*)(ws + 295168);   // 33*1024*32 bf16
  unsigned short* BpkC  = (unsigned short*)(ws + 2457856);  // 33*1024*32 bf16
  unsigned short* S_all = (unsigned short*)(ws + 4620544);  // 2048*16*1024 bf16

  // zero sync counters + Szero (first 33024 bytes of ws)
  k_init_zero<<<64, 256, 0, stream>>>((float*)ws, 8256u);
  // PB / PD (small GEMMs)
  k_pbpd<<<256, 256, 0, stream>>>(proj, imat, dmat, PB, PD);
  // pack B operands into WMMA-fragment order (bf16)
  k_pack<<<4224, 256, 0, stream>>>(smat, PB, BpkA);
  k_pack<<<4224, 256, 0, stream>>>(omat, PD, BpkC);
  // sequential recurrence (persistent cooperative kernel, async LDS staging)
  k_scan<<<NWG, SCAN_THREADS, 0, stream>>>(control, BpkA, Szero, S_all, sync);
  // bulk output GEMM + fused overlap-add (atomic-free scatter)
  k_output<<<8192, 256, 0, stream>>>(control, S_all, Szero, BpkC, out);
}
